// NetEstimator_61289183313931
// MI455X (gfx1250) — compile-verified, matrix operand order checked by reading, not run on hardware
//
#include <hip/hip_runtime.h>
#include <hip/hip_bf16.h>
#include <cstddef>
#include <stdint.h>

// Problem constants (from reference)
#define B_SZ  256
#define T_SZ  512
#define H_SZ  256
#define G_SZ  1024            // 4*H gates
#define BTILE 16
#define NBT   (B_SZ / BTILE)  // 16 batch tiles
#define KT_H  8               // 256 / 32 k-tiles
#define FRAG_ELEMS 512        // 32 lanes * 16 bf16 per WMMA fragment

typedef __attribute__((ext_vector_type(16))) __bf16 bf16x16;
typedef __attribute__((ext_vector_type(8)))  float  f32x8;
typedef __attribute__((ext_vector_type(4)))  unsigned int u32x4;
typedef __attribute__((ext_vector_type(8)))  int i32x8;
typedef __attribute__((ext_vector_type(4)))  int i32x4;

__device__ __forceinline__ unsigned short f2bf(float f) {
  union { float f; unsigned u; } v; v.f = f;
  unsigned r = v.u + 0x7FFFu + ((v.u >> 16) & 1u);
  return (unsigned short)(r >> 16);
}

__device__ __forceinline__ bf16x16 ldfrag(const unsigned short* p) {
  return *reinterpret_cast<const bf16x16*>(p);
}

// ---------------------------------------------------------------------------
// TDM: 1-D async copy of n_dw dwords from global to LDS byte offset lds_off.
// D# group0: count=1, lds_addr, global_addr, type=2 ("image").
// D# group1: data_size=4B, tensor_dim0=n_dw, tensor_dim1=1, tile_dim0=n_dw,
//            tensor_dim0_stride=n_dw. Groups 2/3 zero (<=2D tensor).
// 6-arg builtin form (clang-23 / therock headers).
// Tracked by TENSORcnt; caller pairs with s_wait_tensorcnt + barrier.
// ---------------------------------------------------------------------------
__device__ __forceinline__ void tdm_load_1d(unsigned lds_off, const void* gsrc,
                                            unsigned n_dw) {
  unsigned long long ga = (unsigned long long)(uintptr_t)gsrc;
  u32x4 g0 = { 1u,                                   // count=1, user mode
               lds_off,                               // lds_addr (bytes)
               (unsigned)(ga & 0xFFFFFFFFu),          // global_addr[31:0]
               (unsigned)((ga >> 32) & 0x1FFFFFFu) | (2u << 30) }; // [56:32]|type=2
  i32x8 g1 = { (int)(2u << 16),                          // data_size=4B, mask=0
               (int)((n_dw & 0xFFFFu) << 16),            // tensor_dim0[15:0]
               (int)(((n_dw >> 16) & 0xFFFFu) | (1u << 16)), // td0[31:16]|td1 lo=1
               (int)((n_dw & 0xFFFFu) << 16),            // td1 hi=0 | tile_dim0
               0,                                        // tile_dim1=0, tile_dim2=0
               (int)n_dw,                                // tensor_dim0_stride lo
               0, 0 };
  i32x4 gz4 = { 0, 0, 0, 0 };
  i32x8 gz8 = { 0, 0, 0, 0, 0, 0, 0, 0 };
  __builtin_amdgcn_tensor_load_to_lds(g0, g1, gz4, gz4, gz8, 0);
}

// ---------------------------------------------------------------------------
// Pack an O x K row-major fp32 weight matrix into bf16 WMMA B-fragments.
// ---------------------------------------------------------------------------
__global__ void pack_w_kernel(const float* __restrict__ W, int O, int K,
                              int NT, int KT, unsigned short* __restrict__ out) {
  int total = NT * KT * FRAG_ELEMS;
  for (int idx = blockIdx.x * blockDim.x + threadIdx.x; idx < total;
       idx += gridDim.x * blockDim.x) {
    int f = idx >> 9, r = idx & 511;
    int L = r >> 4, i = r & 15;
    int nt = f / KT, kt = f % KT;
    int n = nt * 16 + (L & 15);
    int k = kt * 32 + ((i & 7) + 8 * (L >> 4) + 16 * (i >> 3));
    float v = (n < O && k < K) ? W[(size_t)n * K + k] : 0.f;
    out[idx] = f2bf(v);
  }
}

// ---------------------------------------------------------------------------
// Pack x_seq (B,T,Ftot) column slice [off, off+F) into A-fragments (K pad 32).
// ---------------------------------------------------------------------------
__global__ void pack_x_kernel(const float* __restrict__ x, int Ftot, int off,
                              int F, unsigned short* __restrict__ out) {
  int total = T_SZ * NBT * FRAG_ELEMS;
  for (int idx = blockIdx.x * blockDim.x + threadIdx.x; idx < total;
       idx += gridDim.x * blockDim.x) {
    int r = idx & 511;
    int f = idx >> 9;             // t*NBT + btile
    int t = f >> 4, bt = f & 15;
    int L = r >> 4, i = r & 15;
    int b = bt * 16 + (L & 15);
    int k = (i & 7) + 8 * (L >> 4) + 16 * (i >> 3);
    float v = (k < F) ? x[((size_t)b * T_SZ + t) * Ftot + off + k] : 0.f;
    out[idx] = f2bf(v);
  }
}

// ---------------------------------------------------------------------------
// Two independent LSTM layers fused (blockIdx.y selects lstm_f / lstm_d).
// Persistent over T. Block = 16 batch rows, 8 wave32 waves.
// Per-step x A-fragments are TDM-prefetched (double buffered) into LDS.
// LDS: gates 64KB + h-frags 8KB + bias 4KB + x staging 2*ktx KB.
// ---------------------------------------------------------------------------
struct LstmPairParams {
  const unsigned short* xf0;  const unsigned short* xf1;
  const unsigned short* wih0; const unsigned short* wih1;
  const unsigned short* whh0; const unsigned short* whh1;
  const float* bih0; const float* bih1;
  const float* bhh0; const float* bhh1;
  unsigned short* out0; unsigned short* out1;
  int ktx;
};

#define LSTM_XBUF_OFF (65536u + 8192u + 4096u)

__global__ void lstm_layer_pair_kernel(LstmPairParams P) {
  extern __shared__ char smem[];
  float*          gates = (float*)smem;                           // 16*1024 fp32
  unsigned short* hfr   = (unsigned short*)(smem + 65536);        // 8*512 bf16
  float*          bsum  = (float*)(smem + 65536 + 8192);          // 1024 fp32
  unsigned short* xbuf  = (unsigned short*)(smem + LSTM_XBUF_OFF);

  const int which = blockIdx.y;
  const unsigned short* xfrags    = which ? P.xf1  : P.xf0;
  const unsigned short* wih_frags = which ? P.wih1 : P.wih0;
  const unsigned short* whh_frags = which ? P.whh1 : P.whh0;
  const float* bih = which ? P.bih1 : P.bih0;
  const float* bhh = which ? P.bhh1 : P.bhh0;
  unsigned short* hfrags_out = which ? P.out1 : P.out0;
  const int ktx = P.ktx;
  const int xus = ktx * FRAG_ELEMS;   // ushorts per step tile
  const int xdw = ktx * 256;          // dwords per step tile

  const int tid  = threadIdx.x;
  const int lane = tid & 31;
  const int wave = tid >> 5;
  const int btile = blockIdx.x;

  for (int i = tid; i < KT_H * FRAG_ELEMS; i += blockDim.x) hfr[i] = 0;
  for (int j = tid; j < G_SZ; j += blockDim.x) bsum[j] = bih[j] + bhh[j];

  float c[16];
#pragma unroll
  for (int i = 0; i < 16; ++i) c[i] = 0.f;
  const int m_own = tid >> 4;
  const int j0    = (tid & 15) * 16;

  // Prologue: TDM prefetch of t=0 A-fragments into staging buffer 0.
  if (wave == 0)
    tdm_load_1d(LSTM_XBUF_OFF,
                xfrags + (size_t)btile * ktx * FRAG_ELEMS, xdw);
  __syncthreads();

  for (int t = 0; t < T_SZ; ++t) {
    // Prefetch t+1 while waiting for t's tile; TENSORcnt discipline.
    if (wave == 0) {
      if (t + 1 < T_SZ) {
        tdm_load_1d(LSTM_XBUF_OFF + (unsigned)(((t + 1) & 1) * xus * 2),
                    xfrags + (size_t)((t + 1) * NBT + btile) * ktx * FRAG_ELEMS,
                    xdw);
        __builtin_amdgcn_s_wait_tensorcnt(1);
      } else {
        __builtin_amdgcn_s_wait_tensorcnt(0);
      }
    }
    __syncthreads();
    const unsigned short* xa = xbuf + (t & 1) * xus + lane * 16;

    // -------- WMMA phase: each wave owns 8 gate n-tiles --------------------
    for (int nn = 0; nn < 8; ++nn) {
      const int nt = wave * 8 + nn;
      f32x8 acc = {};
      const unsigned short* wb =
          wih_frags + ((size_t)nt * ktx) * FRAG_ELEMS + lane * 16;
      for (int kt = 0; kt < ktx; ++kt) {
        bf16x16 a = ldfrag(xa + kt * FRAG_ELEMS);
        bf16x16 b = ldfrag(wb + kt * FRAG_ELEMS);
        acc = __builtin_amdgcn_wmma_f32_16x16x32_bf16(false, a, false, b,
                                                      (short)0, acc, false, false);
      }
      const unsigned short* wh =
          whh_frags + ((size_t)nt * KT_H) * FRAG_ELEMS + lane * 16;
#pragma unroll
      for (int kt = 0; kt < KT_H; ++kt) {
        bf16x16 a = ldfrag(hfr + kt * FRAG_ELEMS + lane * 16);
        bf16x16 b = ldfrag(wh + kt * FRAG_ELEMS);
        acc = __builtin_amdgcn_wmma_f32_16x16x32_bf16(false, a, false, b,
                                                      (short)0, acc, false, false);
      }
      const int n  = nt * 16 + (lane & 15);
      const int mb = (lane >> 4) * 8;
#pragma unroll
      for (int r = 0; r < 8; ++r) gates[(mb + r) * G_SZ + n] = acc[r];
    }
    __syncthreads();
    // -------- gate combine (VALU) -> c,h ; repack h into A-frags ------------
#pragma unroll
    for (int jj = 0; jj < 16; ++jj) {
      const int j = j0 + jj;
      float gi = gates[m_own * G_SZ + j]            + bsum[j];
      float gf = gates[m_own * G_SZ + H_SZ + j]     + bsum[H_SZ + j];
      float gg = gates[m_own * G_SZ + 2 * H_SZ + j] + bsum[2 * H_SZ + j];
      float go = gates[m_own * G_SZ + 3 * H_SZ + j] + bsum[3 * H_SZ + j];
      float si = 1.f / (1.f + __expf(-gi));
      float sf = 1.f / (1.f + __expf(-gf));
      float so = 1.f / (1.f + __expf(-go));
      float cc = sf * c[jj] + si * tanhf(gg);
      c[jj] = cc;
      float h = so * tanhf(cc);
      const int kt   = j >> 5, kp = j & 31;
      const int half = (kp >> 3) & 1;
      const int idx  = (kp & 7) | ((kp >> 4) << 3);
      const int L    = m_own + 16 * half;
      hfr[kt * FRAG_ELEMS + L * 16 + idx] = f2bf(h);
    }
    __syncthreads();
    // -------- coalesced stream of this step's h-frags to global -------------
    {
      uint4* dst = (uint4*)(hfrags_out +
                            ((size_t)(t * NBT + btile) * KT_H) * FRAG_ELEMS);
      const uint4* src = (const uint4*)hfr;
      for (int i = tid; i < (KT_H * FRAG_ELEMS * 2) / 16; i += blockDim.x)
        dst[i] = src[i];
    }
  }
}

// ---------------------------------------------------------------------------
// Scan over T: two WMMA MLPs (259->256 LN leaky -> {9,144} tanh) + 3-dim
// recurrence. f_out/d_out fragments TDM-prefetched into double-buffered LDS.
// ---------------------------------------------------------------------------
struct ScanParams {
  const unsigned short* ffr;
  const unsigned short* dfr;
  const float* u_seq;
  const float* h_prev;
  const unsigned short* w1a; const float* b1a; const float* gwa; const float* gba;
  const unsigned short* w2a; const float* b2a;
  const unsigned short* w1g; const float* b1g; const float* gwg; const float* gbg;
  const unsigned short* w2g; const float* b2g;
  float* out;
};

#define S_OFF_HIDA  0u
#define S_OFF_HIDG  16384u
#define S_OFF_AFRA  32768u
#define S_OFF_AFRG  40960u
#define S_OFF_HFR1  49152u
#define S_OFF_HV    50176u
#define S_OFF_STATS 50432u   // muA[16] ivA[16] muG[16] ivG[16]
#define S_OFF_ADM   50688u   // 16*9 fp32
#define S_OFF_GM    51264u   // 16*144 fp32
#define S_OFF_RED   60480u   // sA/ssA/sG/ssG each 256 fp32
#define S_OFF_FBUF  64576u   // 2 x (8KB f + 8KB d)
#define S_SMEM_SZ   (64576u + 32768u)

__global__ void scan_kernel(ScanParams p) {
  extern __shared__ char smem[];
  float*          hidA = (float*)(smem + S_OFF_HIDA);
  float*          hidG = (float*)(smem + S_OFF_HIDG);
  unsigned short* afrA = (unsigned short*)(smem + S_OFF_AFRA);
  unsigned short* afrG = (unsigned short*)(smem + S_OFF_AFRG);
  unsigned short* hfr1 = (unsigned short*)(smem + S_OFF_HFR1);
  float*          hv   = (float*)(smem + S_OFF_HV);
  float*          muA  = (float*)(smem + S_OFF_STATS);
  float*          ivA  = muA + 16;
  float*          muG  = ivA + 16;
  float*          ivG  = muG + 16;
  float*          adm  = (float*)(smem + S_OFF_ADM);
  float*          gm   = (float*)(smem + S_OFF_GM);
  float*          sA   = (float*)(smem + S_OFF_RED);
  float*          ssA  = sA + 256;
  float*          sG   = ssA + 256;
  float*          ssG  = sG + 256;

  const int tid  = threadIdx.x;
  const int lane = tid & 31;
  const int wave = tid >> 5;
  const int btile = blockIdx.x;

  for (int i = tid; i < FRAG_ELEMS; i += blockDim.x) hfr1[i] = 0;
  __syncthreads();
  if (tid < 16) {
    int b = btile * BTILE + tid;
#pragma unroll
    for (int k = 0; k < 3; ++k) {
      float h0 = p.h_prev[b * 3 + k];
      hv[tid * 3 + k] = h0;
      hfr1[tid * 16 + k] = f2bf(h0);
    }
  }
  // Prologue TDM: stage f/d fragments for t=0 into buffer 0.
  if (wave == 0) {
    tdm_load_1d(S_OFF_FBUF,
                p.ffr + (size_t)btile * KT_H * FRAG_ELEMS, 2048);
    tdm_load_1d(S_OFF_FBUF + 8192u,
                p.dfr + (size_t)btile * KT_H * FRAG_ELEMS, 2048);
  }
  __syncthreads();

  for (int t = 0; t < T_SZ; ++t) {
    if (wave == 0) {
      if (t + 1 < T_SZ) {
        unsigned bo = S_OFF_FBUF + (unsigned)(((t + 1) & 1) * 16384);
        tdm_load_1d(bo,
                    p.ffr + (size_t)((t + 1) * NBT + btile) * KT_H * FRAG_ELEMS,
                    2048);
        tdm_load_1d(bo + 8192u,
                    p.dfr + (size_t)((t + 1) * NBT + btile) * KT_H * FRAG_ELEMS,
                    2048);
        __builtin_amdgcn_s_wait_tensorcnt(2);
      } else {
        __builtin_amdgcn_s_wait_tensorcnt(0);
      }
    }
    __syncthreads();
    const unsigned short* fb =
        (const unsigned short*)(smem + S_OFF_FBUF + (t & 1) * 16384);

    // ---- GEMM1: waves 0-3 -> ad hidden, waves 4-7 -> g hidden ----
    {
      const bool isA = wave < 4;
      const unsigned short* abase = (isA ? fb : fb + 4096) + lane * 16;
      const unsigned short* w1 = isA ? p.w1a : p.w1g;
      float* hid = isA ? hidA : hidG;
      const int w4 = (wave & 3) * 4;
      for (int q = 0; q < 4; ++q) {
        const int nt = w4 + q;
        f32x8 acc = {};
        const unsigned short* wb = w1 + ((size_t)nt * 9) * FRAG_ELEMS + lane * 16;
#pragma unroll
        for (int kt = 0; kt < 8; ++kt) {
          bf16x16 a = ldfrag(abase + kt * FRAG_ELEMS);
          bf16x16 b = ldfrag(wb + kt * FRAG_ELEMS);
          acc = __builtin_amdgcn_wmma_f32_16x16x32_bf16(false, a, false, b,
                                                        (short)0, acc, false, false);
        }
        {  // k-tile 8: h (features 256..258 of concat)
          bf16x16 a = ldfrag(hfr1 + lane * 16);
          bf16x16 b = ldfrag(wb + 8 * FRAG_ELEMS);
          acc = __builtin_amdgcn_wmma_f32_16x16x32_bf16(false, a, false, b,
                                                        (short)0, acc, false, false);
        }
        const int n  = nt * 16 + (lane & 15);
        const int mb = (lane >> 4) * 8;
#pragma unroll
        for (int r = 0; r < 8; ++r) hid[(mb + r) * H_SZ + n] = acc[r];
      }
    }
    __syncthreads();
    // ---- LayerNorm stats: parallel partials then 16-way reduce ----
    {
      const int m = tid >> 4, cnk = tid & 15;
      float s1 = 0.f, q1 = 0.f, s2 = 0.f, q2 = 0.f;
#pragma unroll
      for (int jj = 0; jj < 16; ++jj) {
        int j = cnk * 16 + jj;
        float v = hidA[m * H_SZ + j] + p.b1a[j];
        s1 += v; q1 += v * v;
        float w = hidG[m * H_SZ + j] + p.b1g[j];
        s2 += w; q2 += w * w;
      }
      sA[tid] = s1; ssA[tid] = q1; sG[tid] = s2; ssG[tid] = q2;
    }
    __syncthreads();
    if (tid < 32) {
      const bool isA = tid < 16;
      const int m = tid & 15;
      const float* S = isA ? sA : sG;
      const float* Q = isA ? ssA : ssG;
      float s = 0.f, q = 0.f;
#pragma unroll
      for (int k = 0; k < 16; ++k) { s += S[m * 16 + k]; q += Q[m * 16 + k]; }
      float mu  = s * (1.f / H_SZ);
      float var = q * (1.f / H_SZ) - mu * mu;
      float iv  = rsqrtf(var + 1e-5f);
      if (isA) { muA[m] = mu; ivA[m] = iv; } else { muG[m] = mu; ivG[m] = iv; }
    }
    __syncthreads();
    // ---- normalize + leaky, pack to A-frags for GEMM2 ----
    {
      const int m  = tid >> 4;
      const int jb = (tid & 15) * 16;
#pragma unroll
      for (int jj = 0; jj < 16; ++jj) {
        const int j    = jb + jj;
        const int kt   = j >> 5, kp = j & 31;
        const int half = (kp >> 3) & 1;
        const int idx  = (kp & 7) | ((kp >> 4) << 3);
        const int L    = m + 16 * half;
        const int fo   = kt * FRAG_ELEMS + L * 16 + idx;
        float va = (hidA[m * H_SZ + j] + p.b1a[j] - muA[m]) * ivA[m] * p.gwa[j] + p.gba[j];
        va = va > 0.f ? va : 0.01f * va;
        afrA[fo] = f2bf(va);
        float vg = (hidG[m * H_SZ + j] + p.b1g[j] - muG[m]) * ivG[m] * p.gwg[j] + p.gbg[j];
        vg = vg > 0.f ? vg : 0.01f * vg;
        afrG[fo] = f2bf(vg);
      }
    }
    __syncthreads();
    // ---- GEMM2: tile 0 = ad (O=9), tiles 1..9 = g n-tiles (O=144) ----
    for (int tt = wave; tt < 10; tt += 8) {
      f32x8 acc = {};
      const unsigned short* afr = (tt == 0) ? afrA : afrG;
      const unsigned short* w2  = (tt == 0)
          ? p.w2a : (p.w2g + (size_t)(tt - 1) * 8 * FRAG_ELEMS);
#pragma unroll
      for (int kt = 0; kt < 8; ++kt) {
        bf16x16 a = ldfrag(afr + kt * FRAG_ELEMS + lane * 16);
        bf16x16 b = ldfrag(w2 + kt * FRAG_ELEMS + lane * 16);
        acc = __builtin_amdgcn_wmma_f32_16x16x32_bf16(false, a, false, b,
                                                      (short)0, acc, false, false);
      }
      const int nl = lane & 15;
      const int mb = (lane >> 4) * 8;
      if (tt == 0) {
        if (nl < 9) {
#pragma unroll
          for (int r = 0; r < 8; ++r)
            adm[(mb + r) * 9 + nl] = tanhf(acc[r] + p.b2a[nl]);
        }
      } else {
        const int ng = (tt - 1) * 16 + nl;
        if (ng < 144) {
#pragma unroll
          for (int r = 0; r < 8; ++r)
            gm[(mb + r) * 144 + ng] = tanhf(acc[r] + p.b2g[ng]);
        }
      }
    }
    __syncthreads();
    // ---- tiny recurrence: gu = g@u; h_next = ad@h + fold(gu) ----
    if (tid < 16) {
      const int m = tid;
      const int b = btile * BTILE + m;
      float u[12];
#pragma unroll
      for (int j = 0; j < 12; ++j) u[j] = p.u_seq[((size_t)b * T_SZ + t) * 12 + j];
      float hl[3] = { hv[m * 3 + 0], hv[m * 3 + 1], hv[m * 3 + 2] };
      float sg[3] = { 0.f, 0.f, 0.f };
#pragma unroll
      for (int i = 0; i < 12; ++i) {
        float s = 0.f;
#pragma unroll
        for (int j = 0; j < 12; ++j) s += gm[m * 144 + i * 12 + j] * u[j];
        sg[i % 3] += s;
      }
#pragma unroll
      for (int k = 0; k < 3; ++k) {
        float hn = adm[m * 9 + k * 3 + 0] * hl[0] +
                   adm[m * 9 + k * 3 + 1] * hl[1] +
                   adm[m * 9 + k * 3 + 2] * hl[2] + sg[k];
        p.out[((size_t)b * T_SZ + t) * 3 + k] = hn;
        hv[m * 3 + k] = hn;
        hfr1[m * 16 + k] = f2bf(hn);
      }
    }
    __syncthreads();
  }
}

// ---------------------------------------------------------------------------
// Host orchestration. Input order: x_seq, u_seq, h_prev, then params pytree
// with sorted dict keys (ad, g, lstm_d, lstm_f).
// ---------------------------------------------------------------------------
extern "C" void kernel_launch(void* const* d_in, const int* in_sizes, int n_in,
                              void* d_out, int out_size, void* d_ws, size_t ws_size,
                              hipStream_t stream) {
  (void)in_sizes; (void)n_in; (void)out_size; (void)ws_size;
  const float* x_seq  = (const float*)d_in[0];
  const float* u_seq  = (const float*)d_in[1];
  const float* h_prev = (const float*)d_in[2];
  const float* W1a = (const float*)d_in[3];
  const float* b1a = (const float*)d_in[4];
  const float* gwa = (const float*)d_in[5];
  const float* gba = (const float*)d_in[6];
  const float* W2a = (const float*)d_in[7];
  const float* b2a = (const float*)d_in[8];
  const float* W1g = (const float*)d_in[9];
  const float* b1g = (const float*)d_in[10];
  const float* gwg = (const float*)d_in[11];
  const float* gbg = (const float*)d_in[12];
  const float* W2g = (const float*)d_in[13];
  const float* b2g = (const float*)d_in[14];
  const float* d0_Wih = (const float*)d_in[15];
  const float* d0_Whh = (const float*)d_in[16];
  const float* d0_bih = (const float*)d_in[17];
  const float* d0_bhh = (const float*)d_in[18];
  const float* d1_Wih = (const float*)d_in[19];
  const float* d1_Whh = (const float*)d_in[20];
  const float* d1_bih = (const float*)d_in[21];
  const float* d1_bhh = (const float*)d_in[22];
  const float* f0_Wih = (const float*)d_in[23];
  const float* f0_Whh = (const float*)d_in[24];
  const float* f0_bih = (const float*)d_in[25];
  const float* f0_bhh = (const float*)d_in[26];
  const float* f1_Wih = (const float*)d_in[27];
  const float* f1_Whh = (const float*)d_in[28];
  const float* f1_bih = (const float*)d_in[29];
  const float* f1_bhh = (const float*)d_in[30];

  char* ws = (char*)d_ws;
  size_t off = 0;
  auto take = [&](size_t bytes) -> char* {
    char* pp = ws + off;
    off += (bytes + 255) & ~(size_t)255;
    return pp;
  };
  const size_t FRAG_B = FRAG_ELEMS * 2;
  unsigned short* pf0_wih = (unsigned short*)take(64ull * 1 * FRAG_B);
  unsigned short* pf0_whh = (unsigned short*)take(64ull * 8 * FRAG_B);
  unsigned short* pf1_wih = (unsigned short*)take(64ull * 8 * FRAG_B);
  unsigned short* pf1_whh = (unsigned short*)take(64ull * 8 * FRAG_B);
  unsigned short* pd0_wih = (unsigned short*)take(64ull * 1 * FRAG_B);
  unsigned short* pd0_whh = (unsigned short*)take(64ull * 8 * FRAG_B);
  unsigned short* pd1_wih = (unsigned short*)take(64ull * 8 * FRAG_B);
  unsigned short* pd1_whh = (unsigned short*)take(64ull * 8 * FRAG_B);
  unsigned short* pw1a = (unsigned short*)take(16ull * 9 * FRAG_B);
  unsigned short* pw2a = (unsigned short*)take(1ull * 8 * FRAG_B);
  unsigned short* pw1g = (unsigned short*)take(16ull * 9 * FRAG_B);
  unsigned short* pw2g = (unsigned short*)take(9ull * 8 * FRAG_B);
  unsigned short* pxf   = (unsigned short*)take((size_t)T_SZ * NBT * 1 * FRAG_B);
  unsigned short* pxd   = (unsigned short*)take((size_t)T_SZ * NBT * 1 * FRAG_B);
  unsigned short* pmidF = (unsigned short*)take((size_t)T_SZ * NBT * KT_H * FRAG_B);
  unsigned short* pmidD = (unsigned short*)take((size_t)T_SZ * NBT * KT_H * FRAG_B);
  unsigned short* pffr  = (unsigned short*)take((size_t)T_SZ * NBT * KT_H * FRAG_B);
  unsigned short* pdfr  = (unsigned short*)take((size_t)T_SZ * NBT * KT_H * FRAG_B);

  dim3 blk(256);
  auto packw = [&](const float* W, int O, int K, int NT, int KT, unsigned short* o) {
    int total = NT * KT * FRAG_ELEMS;
    int grid = (total + 255) / 256;
    pack_w_kernel<<<grid, blk, 0, stream>>>(W, O, K, NT, KT, o);
  };
  packw(f0_Wih, G_SZ, 9,    64, 1, pf0_wih);
  packw(f0_Whh, G_SZ, H_SZ, 64, 8, pf0_whh);
  packw(f1_Wih, G_SZ, H_SZ, 64, 8, pf1_wih);
  packw(f1_Whh, G_SZ, H_SZ, 64, 8, pf1_whh);
  packw(d0_Wih, G_SZ, 12,   64, 1, pd0_wih);
  packw(d0_Whh, G_SZ, H_SZ, 64, 8, pd0_whh);
  packw(d1_Wih, G_SZ, H_SZ, 64, 8, pd1_wih);
  packw(d1_Whh, G_SZ, H_SZ, 64, 8, pd1_whh);
  packw(W1a, 256, 259, 16, 9, pw1a);
  packw(W2a, 9,   256, 1,  8, pw2a);
  packw(W1g, 256, 259, 16, 9, pw1g);
  packw(W2g, 144, 256, 9,  8, pw2g);
  {
    int total = T_SZ * NBT * FRAG_ELEMS;
    int grid = (total + 255) / 256;
    pack_x_kernel<<<grid, blk, 0, stream>>>(x_seq, 21, 0, 9,  pxf);
    pack_x_kernel<<<grid, blk, 0, stream>>>(x_seq, 21, 9, 12, pxd);
  }

  // Fused LSTM pairs: layer 0 of both LSTMs concurrently, then layer 1.
  {
    LstmPairParams L0;
    L0.xf0 = pxf;      L0.xf1 = pxd;
    L0.wih0 = pf0_wih; L0.wih1 = pd0_wih;
    L0.whh0 = pf0_whh; L0.whh1 = pd0_whh;
    L0.bih0 = f0_bih;  L0.bih1 = d0_bih;
    L0.bhh0 = f0_bhh;  L0.bhh1 = d0_bhh;
    L0.out0 = pmidF;   L0.out1 = pmidD;
    L0.ktx = 1;
    size_t smem0 = LSTM_XBUF_OFF + 2ull * 1 * 1024;
    lstm_layer_pair_kernel<<<dim3(NBT, 2), blk, smem0, stream>>>(L0);

    LstmPairParams L1;
    L1.xf0 = pmidF;    L1.xf1 = pmidD;
    L1.wih0 = pf1_wih; L1.wih1 = pd1_wih;
    L1.whh0 = pf1_whh; L1.whh1 = pd1_whh;
    L1.bih0 = f1_bih;  L1.bih1 = d1_bih;
    L1.bhh0 = f1_bhh;  L1.bhh1 = d1_bhh;
    L1.out0 = pffr;    L1.out1 = pdfr;
    L1.ktx = 8;
    size_t smem1 = LSTM_XBUF_OFF + 2ull * 8 * 1024;
    lstm_layer_pair_kernel<<<dim3(NBT, 2), blk, smem1, stream>>>(L1);
  }

  // Scan with the two WMMA MLPs.
  ScanParams sp;
  sp.ffr = pffr; sp.dfr = pdfr; sp.u_seq = u_seq; sp.h_prev = h_prev;
  sp.w1a = pw1a; sp.b1a = b1a; sp.gwa = gwa; sp.gba = gba;
  sp.w2a = pw2a; sp.b2a = b2a;
  sp.w1g = pw1g; sp.b1g = b1g; sp.gwg = gwg; sp.gbg = gbg;
  sp.w2g = pw2g; sp.b2g = b2g;
  sp.out = (float*)d_out;
  scan_kernel<<<NBT, blk, S_SMEM_SZ, stream>>>(sp);
}
